// ConditionalAdapter_43611097924302
// MI455X (gfx1250) — matrix operand order repaired and checked by reading, not compiled, not run on hardware
//
#include <hip/hip_runtime.h>
#include <stddef.h>

// ---------------------------------------------------------------------------
// ConditionalAdapter fused kernel for MI455X (gfx1250, wave32, WMMA bf16)
//
// Math (exact refactor of the reference):
//   dg = cond @ down_gamma   [B,H]      db = cond @ down_beta  [B,H]
//   ug = cond @ up_gamma     [B,D]      ub = cond @ up_beta    [B,D]
//   sX[r] = sum_d X[r,d]
//   act   = relu(dg*(X @ W_down) + db*sX)
//   sA[r] = sum_h act[r,h]
//   out   = ug*(act @ U) + ub*sA + X
//
// GEMMs: bf16 hi/lo split (3x v_wmma_f32_16x16x32_bf16 per K-step tile),
// f32 accumulation (~16 mantissa bits). Weights pre-converted/pre-transposed
// to bf16 once; hot-loop B staging is a pure async global->LDS copy.
// ---------------------------------------------------------------------------

#define Bb   8
#define Ss   4096
#define Dd   1024
#define Hh   256
#define Cc   512
#define ROWS (Bb * Ss)   // 32768

typedef __attribute__((ext_vector_type(16))) __bf16        bf16x16;
typedef __attribute__((ext_vector_type(8)))  float         f32x8;
typedef __attribute__((ext_vector_type(4)))  float         f32x4;
typedef __attribute__((ext_vector_type(4)))  unsigned int  u32x4;

union FragU { u32x4 q[2]; bf16x16 v; };

__device__ __forceinline__ unsigned short f32_to_bf16(float f) {
  unsigned int u = __float_as_uint(f);
  u += 0x7fffu + ((u >> 16) & 1u);        // round-to-nearest-even
  return (unsigned short)(u >> 16);
}
__device__ __forceinline__ float bf16_to_f32(unsigned short h) {
  return __uint_as_float(((unsigned int)h) << 16);
}

// CDNA5 async global->LDS 16B copy (ASYNCcnt-tracked). LDS operand is the low
// 32 bits of the flat shared-aperture address (== LDS byte offset).
__device__ __forceinline__ void async_copy16(const void* g, void* l) {
  asm volatile("global_load_async_to_lds_b128 %0, %1, off"
               :: "v"((unsigned int)(unsigned long long)l),
                  "v"((unsigned long long)g)
               : "memory");
}
__device__ __forceinline__ void wait_async0() {
  asm volatile("s_wait_asynccnt 0x0" ::: "memory");
}

// A fragment (16x32 bf16, row-major LDS [m][k], stride lda elems):
// lanes 0-15: M=lane, K = 0..7 & 16..23 ; lanes 16-31: M=lane-16, K = 8..15 & 24..31
__device__ __forceinline__ bf16x16 ldsA(const unsigned short* base, int lda,
                                        int m0, int k0, int lane) {
  const int l16 = lane & 15, lh = lane >> 4;
  const unsigned short* p = base + (size_t)(m0 + l16) * lda + k0 + lh * 8;
  FragU f;
  f.q[0] = *(const u32x4*)(p);
  f.q[1] = *(const u32x4*)(p + 16);
  return f.v;
}

// B fragment (32x16 bf16, LDS staged transposed [n][k], stride ldb elems):
// lanes 0-15: N=lane, K = 0..7 & 16..23 ; lanes 16-31: N=lane-16, K = 8..15 & 24..31
__device__ __forceinline__ bf16x16 ldsB(const unsigned short* base, int ldb,
                                        int n0, int lane) {
  const int l16 = lane & 15, lh = lane >> 4;
  const unsigned short* p = base + (size_t)(n0 + l16) * ldb + lh * 8;
  FragU f;
  f.q[0] = *(const u32x4*)(p);
  f.q[1] = *(const u32x4*)(p + 16);
  return f.v;
}

__device__ __forceinline__ f32x8 wmma_bf16(bf16x16 a, bf16x16 b, f32x8 c) {
  return __builtin_amdgcn_wmma_f32_16x16x32_bf16(false, a, false, b,
                                                 (short)0, c, false, false);
}

// ---------------------------------------------------------------------------
// Kernel 1: modulation coefficients dg/db [B,H], ug/ub [B,D]  (tiny GEMMs)
// ---------------------------------------------------------------------------
__global__ __launch_bounds__(256) void modulators_kernel(
    const float* __restrict__ cond,
    const float* __restrict__ down_gamma, const float* __restrict__ down_beta,
    const float* __restrict__ up_gamma,   const float* __restrict__ up_beta,
    float* __restrict__ dg, float* __restrict__ db,
    float* __restrict__ ug, float* __restrict__ ub)
{
  const int g = blockIdx.x * 256 + threadIdx.x;   // 0 .. 20479
  const float* gm; float* outp; int ncol, idx;
  if      (g < 2048)  { gm = down_gamma; outp = dg; ncol = Hh; idx = g; }
  else if (g < 4096)  { gm = down_beta;  outp = db; ncol = Hh; idx = g - 2048; }
  else if (g < 12288) { gm = up_gamma;   outp = ug; ncol = Dd; idx = g - 4096; }
  else                { gm = up_beta;    outp = ub; ncol = Dd; idx = g - 12288; }
  const int bI = idx / ncol, col = idx % ncol;
  const float* cp = cond + (size_t)bI * Cc;
  float s = 0.f;
  #pragma unroll 4
  for (int c = 0; c < Cc; ++c)
    s = fmaf(cp[c], gm[(size_t)c * ncol + col], s);
  outp[idx] = s;
}

// ---------------------------------------------------------------------------
// Kernel 2: one-time weight transpose + fp32 -> bf16 hi/lo split.
// blocks 0..255:   Wd [1024][256] -> WdT_hi/lo [256][1024]
// blocks 256..511: Up [256][1024] -> UpT_hi/lo [1024][256]
// ---------------------------------------------------------------------------
__global__ __launch_bounds__(256) void transpose_convert_kernel(
    const float* __restrict__ Wd, const float* __restrict__ Up,
    unsigned short* __restrict__ WdTHi, unsigned short* __restrict__ WdTLo,
    unsigned short* __restrict__ UpTHi, unsigned short* __restrict__ UpTLo)
{
  const float* src; unsigned short *dH, *dL; int R, C, tile;
  if (blockIdx.x < 256) { src = Wd; dH = WdTHi; dL = WdTLo; R = Dd; C = Hh; tile = blockIdx.x; }
  else                  { src = Up; dH = UpTHi; dL = UpTLo; R = Hh; C = Dd; tile = blockIdx.x - 256; }
  const int tilesC = C >> 5;
  const int tr = tile / tilesC, tc = tile % tilesC;

  __shared__ unsigned short lH[32][33], lL[32][33];
  const int tid = threadIdx.x;
  const int rl  = tid >> 5;            // 0..7
  const int cl  = tid & 31;

  #pragma unroll
  for (int j = 0; j < 4; ++j) {
    const int r = rl + j * 8;
    const float v = src[(size_t)(tr * 32 + r) * C + tc * 32 + cl];
    const unsigned short h = f32_to_bf16(v);
    lH[r][cl] = h;
    lL[r][cl] = f32_to_bf16(v - bf16_to_f32(h));
  }
  __syncthreads();
  #pragma unroll
  for (int j = 0; j < 2; ++j) {
    const int u  = tid + j * 256;      // 0..511
    const int c  = u >> 4;             // 0..31
    const int rp = (u & 15) * 2;       // 0..30 (even)
    const unsigned int hv = (unsigned int)lH[rp][c] | ((unsigned int)lH[rp + 1][c] << 16);
    const unsigned int lv = (unsigned int)lL[rp][c] | ((unsigned int)lL[rp + 1][c] << 16);
    const size_t off = (size_t)(tc * 32 + c) * R + tr * 32 + rp;
    *(unsigned int*)&dH[off] = hv;
    *(unsigned int*)&dL[off] = lv;
  }
}

// ---------------------------------------------------------------------------
// Kernel 3: fused dual-GEMM adapter. One block = 64 rows, 8 waves (2x4 grid).
// Double-buffered LDS stages; B tiles arrive via async global->LDS copies.
// ---------------------------------------------------------------------------
__global__ __launch_bounds__(256, 1) void fused_adapter_kernel(
    const float* __restrict__ X,            // [ROWS, D] fp32
    const unsigned short* __restrict__ WdTHi, const unsigned short* __restrict__ WdTLo, // [H][D] bf16
    const unsigned short* __restrict__ UpTHi, const unsigned short* __restrict__ UpTLo, // [D][H] bf16
    const float* __restrict__ dg, const float* __restrict__ db,   // [B,H]
    const float* __restrict__ ug, const float* __restrict__ ub,   // [B,D]
    float* __restrict__ Out)                // [ROWS, D] fp32
{
  __shared__ __align__(16) unsigned short aHiS[2][64][40];    // 10 KB  X tile hi (dbl buf)
  __shared__ __align__(16) unsigned short aLoS[2][64][40];    // 10 KB  X tile lo
  __shared__ __align__(16) unsigned short bHiS[2][256][40];   // 40 KB  W/U tile hi ([n][k], dbl buf)
  __shared__ __align__(16) unsigned short bLoS[2][256][40];   // 40 KB  W/U tile lo
  __shared__ __align__(16) unsigned short actHi[64][264];     // 33 KB  act hi (row-major [m][h])
  __shared__ __align__(16) unsigned short actLo[64][264];     // 33 KB  act lo
  __shared__ __align__(16) float sumX[64];
  __shared__ __align__(16) float sumA[64];
  __shared__ __align__(16) float redBuf[64][4];

  const int tid  = threadIdx.x;
  const int lane = tid & 31;
  const int wave = tid >> 5;
  const int wm   = wave >> 2;          // 0..1  (M tiles of 32 rows)
  const int wn   = wave & 3;           // 0..3  (N tiles of 64 cols)
  const int l16  = lane & 15;
  const int lh   = lane >> 4;

  const int m0 = blockIdx.x * 64;      // 512 blocks; 64 | 4096 -> batch-uniform
  const int bI = m0 >> 12;             // batch index

  const int xr  = tid >> 2;            // X stage: row 0..63
  const int xc0 = (tid & 3) * 8;       // X stage: 8-col group
  const int bc  = (tid & 3) * 8;       // B stage: 16B chunk within 32-k row
  const int bn  = tid >> 2;            // B stage: base row 0..63 (+64*j)

  f32x8 acc[2][4];
  float sX = 0.0f;

  // ---- staging helpers (lambdas keep shared-state capture simple) --------
  auto stage_X = [&](int p, int k0) {
    const float* xp = X + (size_t)(m0 + xr) * Dd + k0 + xc0;
    f32x4 x0 = *(const f32x4*)(xp);
    f32x4 x1 = *(const f32x4*)(xp + 4);
    float e[8];
    #pragma unroll
    for (int i = 0; i < 4; ++i) { e[i] = x0[i]; e[4 + i] = x1[i]; }
    u32x4 hv, lv;
    #pragma unroll
    for (int j = 0; j < 4; ++j) {
      const float a0 = e[2 * j], a1 = e[2 * j + 1];
      sX += a0 + a1;
      const unsigned short h0 = f32_to_bf16(a0), h1 = f32_to_bf16(a1);
      const unsigned short g0 = f32_to_bf16(a0 - bf16_to_f32(h0));
      const unsigned short g1 = f32_to_bf16(a1 - bf16_to_f32(h1));
      hv[j] = (unsigned int)h0 | ((unsigned int)h1 << 16);
      lv[j] = (unsigned int)g0 | ((unsigned int)g1 << 16);
    }
    *(u32x4*)&aHiS[p][xr][xc0] = hv;
    *(u32x4*)&aLoS[p][xr][xc0] = lv;
  };

  auto stage_B = [&](int p, const unsigned short* srcHi,
                     const unsigned short* srcLo, int ldK, int k0) {
    #pragma unroll
    for (int j = 0; j < 4; ++j) {
      const int n = bn + 64 * j;
      async_copy16(srcHi + (size_t)n * ldK + k0 + bc, &bHiS[p][n][bc]);
      async_copy16(srcLo + (size_t)n * ldK + k0 + bc, &bLoS[p][n][bc]);
    }
  };

  // =========================== GEMM 1: X @ W_down =========================
  #pragma unroll
  for (int mt = 0; mt < 2; ++mt)
    #pragma unroll
    for (int nt = 0; nt < 4; ++nt) acc[mt][nt] = 0.0f;

  stage_X(0, 0);
  stage_B(0, WdTHi, WdTLo, Dd, 0);

  for (int kt = 0; kt < 32; ++kt) {
    const int p = kt & 1;
    wait_async0();
    __syncthreads();
    if (kt + 1 < 32) {
      stage_X(1 - p, (kt + 1) * 32);
      stage_B(1 - p, WdTHi, WdTLo, Dd, (kt + 1) * 32);
    }
    bf16x16 aH[2], aL[2];
    #pragma unroll
    for (int mt = 0; mt < 2; ++mt) {
      aH[mt] = ldsA(&aHiS[p][0][0], 40, wm * 32 + mt * 16, 0, lane);
      aL[mt] = ldsA(&aLoS[p][0][0], 40, wm * 32 + mt * 16, 0, lane);
    }
    #pragma unroll
    for (int nt = 0; nt < 4; ++nt) {
      const bf16x16 bH = ldsB(&bHiS[p][0][0], 40, wn * 64 + nt * 16, lane);
      const bf16x16 bL = ldsB(&bLoS[p][0][0], 40, wn * 64 + nt * 16, lane);
      #pragma unroll
      for (int mt = 0; mt < 2; ++mt) {
        acc[mt][nt] = wmma_bf16(aH[mt], bH, acc[mt][nt]);   // hi*hi
        acc[mt][nt] = wmma_bf16(aH[mt], bL, acc[mt][nt]);   // hi*lo
        acc[mt][nt] = wmma_bf16(aL[mt], bH, acc[mt][nt]);   // lo*hi
      }
    }
  }

  // rowsum(X): 4 partials per row -> sumX
  redBuf[xr][tid & 3] = sX;
  __syncthreads();
  if (tid < 64)
    sumX[tid] = redBuf[tid][0] + redBuf[tid][1] + redBuf[tid][2] + redBuf[tid][3];
  __syncthreads();

  // ============== epilogue 1: act = relu(dg*Y + db*sumX) -> LDS ===========
  #pragma unroll
  for (int mt = 0; mt < 2; ++mt) {
    const int mbase = wm * 32 + mt * 16 + 8 * lh;
    const f32x4 s0 = *(const f32x4*)&sumX[mbase];
    const f32x4 s1 = *(const f32x4*)&sumX[mbase + 4];
    #pragma unroll
    for (int nt = 0; nt < 4; ++nt) {
      const int n = wn * 64 + nt * 16 + l16;               // H index
      const float dgv = dg[bI * Hh + n];
      const float dbv = db[bI * Hh + n];
      #pragma unroll
      for (int i = 0; i < 8; ++i) {
        const float sx = (i < 4) ? s0[i] : s1[i - 4];
        float v = dgv * acc[mt][nt][i] + dbv * sx;
        v = v > 0.f ? v : 0.f;                             // relu
        const unsigned short h = f32_to_bf16(v);
        const unsigned short l = f32_to_bf16(v - bf16_to_f32(h));
        actHi[mbase + i][n] = h;
        actLo[mbase + i][n] = l;
      }
    }
  }
  __syncthreads();

  // rowsum(act)
  {
    const int c0 = (tid & 3) * 64;
    float s = 0.f;
    #pragma unroll 8
    for (int c = 0; c < 64; c += 2) {
      const unsigned int ph = *(const unsigned int*)&actHi[xr][c0 + c];
      const unsigned int pl = *(const unsigned int*)&actLo[xr][c0 + c];
      s += bf16_to_f32((unsigned short)(ph & 0xffffu)) + bf16_to_f32((unsigned short)(ph >> 16));
      s += bf16_to_f32((unsigned short)(pl & 0xffffu)) + bf16_to_f32((unsigned short)(pl >> 16));
    }
    redBuf[xr][tid & 3] = s;
  }
  __syncthreads();
  if (tid < 64)
    sumA[tid] = redBuf[tid][0] + redBuf[tid][1] + redBuf[tid][2] + redBuf[tid][3];
  __syncthreads();

  // ======================= GEMM 2: act @ U, 4 N-chunks ====================
  for (int nc = 0; nc < 4; ++nc) {
    const unsigned short* uHi = UpTHi + (size_t)(nc * 256) * Hh;  // rows d of [D][H]
    const unsigned short* uLo = UpTLo + (size_t)(nc * 256) * Hh;

    #pragma unroll
    for (int mt = 0; mt < 2; ++mt)
      #pragma unroll
      for (int nt = 0; nt < 4; ++nt) acc[mt][nt] = 0.0f;

    stage_B(0, uHi, uLo, Hh, 0);

    for (int kt = 0; kt < 8; ++kt) {
      const int p = kt & 1;
      const int kk = kt * 32;
      wait_async0();
      __syncthreads();
      if (kt + 1 < 8) stage_B(1 - p, uHi, uLo, Hh, (kt + 1) * 32);

      bf16x16 aH[2], aL[2];
      #pragma unroll
      for (int mt = 0; mt < 2; ++mt) {
        aH[mt] = ldsA(&actHi[0][0], 264, wm * 32 + mt * 16, kk, lane);
        aL[mt] = ldsA(&actLo[0][0], 264, wm * 32 + mt * 16, kk, lane);
      }
      #pragma unroll
      for (int nt = 0; nt < 4; ++nt) {
        const bf16x16 bH = ldsB(&bHiS[p][0][0], 40, wn * 64 + nt * 16, lane);
        const bf16x16 bL = ldsB(&bLoS[p][0][0], 40, wn * 64 + nt * 16, lane);
        #pragma unroll
        for (int mt = 0; mt < 2; ++mt) {
          acc[mt][nt] = wmma_bf16(aH[mt], bH, acc[mt][nt]);
          acc[mt][nt] = wmma_bf16(aH[mt], bL, acc[mt][nt]);
          acc[mt][nt] = wmma_bf16(aL[mt], bH, acc[mt][nt]);
        }
      }
    }

    // epilogue 2: out = ug*acc + ub*sumA + X (residual)
    #pragma unroll
    for (int mt = 0; mt < 2; ++mt) {
      const int mbase = wm * 32 + mt * 16 + 8 * lh;
      const f32x4 s0 = *(const f32x4*)&sumA[mbase];
      const f32x4 s1 = *(const f32x4*)&sumA[mbase + 4];
      #pragma unroll
      for (int nt = 0; nt < 4; ++nt) {
        const int n = nc * 256 + wn * 64 + nt * 16 + l16;  // D index
        const float ugv = ug[bI * Dd + n];
        const float ubv = ub[bI * Dd + n];
        #pragma unroll
        for (int i = 0; i < 8; ++i) {
          const size_t row = (size_t)(m0 + mbase + i);
          const float sa = (i < 4) ? s0[i] : s1[i - 4];
          const float xres = X[row * Dd + n];
          Out[row * Dd + n] = ugv * acc[mt][nt][i] + ubv * sa + xres;
        }
      }
    }
  }
}

// ---------------------------------------------------------------------------
extern "C" void kernel_launch(void* const* d_in, const int* in_sizes, int n_in,
                              void* d_out, int out_size, void* d_ws, size_t ws_size,
                              hipStream_t stream)
{
  (void)in_sizes; (void)n_in; (void)out_size; (void)ws_size;

  const float* hidden       = (const float*)d_in[0];  // [B,S,D]
  const float* conditions   = (const float*)d_in[1];  // [B,C]
  const float* down_project = (const float*)d_in[2];  // [D,H]
  const float* down_gamma   = (const float*)d_in[3];  // [C,H]
  const float* down_beta    = (const float*)d_in[4];  // [C,H]
  const float* up_project   = (const float*)d_in[5];  // [H,D]
  const float* up_gamma     = (const float*)d_in[6];  // [C,D]
  const float* up_beta      = (const float*)d_in[7];  // [C,D]
  float* out = (float*)d_out;

  float* wsf = (float*)d_ws;
  float* dg = wsf;                 // [B,H]  2048
  float* db = wsf + 2048;          // [B,H]  2048
  float* ug = wsf + 4096;          // [B,D]  8192
  float* ub = wsf + 12288;         // [B,D]  8192

  unsigned short* wsu = (unsigned short*)((char*)d_ws + 20480 * sizeof(float));
  unsigned short* WdTHi = wsu;                   // [H][D] 256x1024 bf16
  unsigned short* WdTLo = wsu + 1 * 262144;
  unsigned short* UpTHi = wsu + 2 * 262144;      // [D][H] 1024x256 bf16
  unsigned short* UpTLo = wsu + 3 * 262144;

  hipLaunchKernelGGL(modulators_kernel, dim3(80), dim3(256), 0, stream,
                     conditions, down_gamma, down_beta, up_gamma, up_beta,
                     dg, db, ug, ub);

  hipLaunchKernelGGL(transpose_convert_kernel, dim3(512), dim3(256), 0, stream,
                     down_project, up_project, WdTHi, WdTLo, UpTHi, UpTLo);

  hipLaunchKernelGGL(fused_adapter_kernel, dim3(ROWS / 64), dim3(256), 0, stream,
                     hidden, WdTHi, WdTLo, UpTHi, UpTLo, dg, db, ug, ub, out);
}
